// GRUSeq2SeqWithAttentionWithBatzzang_20091857011059
// MI455X (gfx1250) — compile-verified, matrix operand order checked by reading, not run on hardware
//
#include <hip/hip_runtime.h>
#include <hip/hip_bf16.h>

typedef __attribute__((ext_vector_type(16))) _Float16 v16h;
typedef __attribute__((ext_vector_type(8)))  float    v8f;

#define HDIM 512
#define TH3  1536
#define SEQ  256
#define BB   8
#define VOC  32000
#define NROW 2048          // B*SEQ
#define VSPLIT 10
#define VCHUNK (VOC / VSPLIT)   // 3200 = 25 chunks of 128

// ---------------------------------------------------------------------------
// WMMA fragment helpers (16x16x32 f16, wave32). Per CDNA5 ISA:
// 16-bit A (16x32): lane m=lane&15, hi=lane>>4; element e -> k = (e&8)*2 + hi*8 + (e&7)
// i.e. two contiguous 8-half runs at k0+hi*8 and k0+16+hi*8 (vectorizes to b128).
// B (32x16) symmetric with n=lane&15 when B is stored [N,K] K-contiguous.
// C/D f32: lane n=lane&15, VGPR r -> row m = r + hi*8.
// ---------------------------------------------------------------------------
__device__ __forceinline__ v16h load_frag(const _Float16* base, int ld, int lane, int k0) {
    int m  = lane & 15;
    int hi = (lane >> 4) & 1;
    const _Float16* p = base + (size_t)m * ld + k0 + hi * 8;
    v16h f;
#pragma unroll
    for (int e = 0; e < 8; ++e) { f[e] = p[e]; f[e + 8] = p[16 + e]; }
    return f;
}

__device__ __forceinline__ v8f wmma_f16(v16h a, v16h b, v8f c) {
    return __builtin_amdgcn_wmma_f32_16x16x32_f16(false, a, false, b, (short)0, c, false, false);
}

// ---------------------------------------------------------------------------
// elementwise conversion / layout kernels
// ---------------------------------------------------------------------------
__global__ void cvt_f32_f16_kernel(_Float16* dst, const float* src, int n) {
    for (int i = blockIdx.x * blockDim.x + threadIdx.x; i < n; i += gridDim.x * blockDim.x)
        dst[i] = (_Float16)src[i];
}

// dst[c*rows + r] = src[r*cols + c]  (transpose to [cols, rows], K-contiguous B layout)
__global__ void transpose_f32_f16_kernel(_Float16* dst, const float* src, int rows, int cols) {
    int n = rows * cols;
    for (int i = blockIdx.x * blockDim.x + threadIdx.x; i < n; i += gridDim.x * blockDim.x) {
        int r = i / cols, c = i - r * cols;
        dst[(size_t)c * rows + r] = (_Float16)src[(size_t)r * cols + c];
    }
}

// gather embeddings for encoder input and teacher-forced decoder input
__global__ void gather_emb_kernel(_Float16* enc_emb, _Float16* dec_emb,
                                  const float* embedding,
                                  const int* input_seq, const int* target_seq) {
    int n = BB * SEQ * HDIM;
    for (int i = blockIdx.x * blockDim.x + threadIdx.x; i < n; i += gridDim.x * blockDim.x) {
        int h  = i & (HDIM - 1);
        int bs = i >> 9;                 // b*SEQ + s
        int s  = bs & (SEQ - 1);
        int etok = input_seq[bs];
        int dtok = (s == 0) ? 1 /*SOS*/ : target_seq[bs - 1];
        enc_emb[i] = (_Float16)embedding[(size_t)etok * HDIM + h];
        dec_emb[i] = (_Float16)embedding[(size_t)dtok * HDIM + h];
    }
}

// enc_out = enc_f + enc_b (already time-aligned); emit f16 + [B,H,S] transposed f16
// and dec_out f16.
__global__ void combine_kernel(const float* enc_f, const float* enc_b, const float* dec_out,
                               float* enc_out, _Float16* enc_out16, _Float16* enc_outT16,
                               _Float16* dec_out16) {
    int n = BB * SEQ * HDIM;
    for (int i = blockIdx.x * blockDim.x + threadIdx.x; i < n; i += gridDim.x * blockDim.x) {
        int h  = i & (HDIM - 1);
        int bs = i >> 9;
        int s  = bs & (SEQ - 1);
        int b  = bs >> 8;
        float v = enc_f[i] + enc_b[i];
        enc_out[i]   = v;
        enc_out16[i] = (_Float16)v;
        enc_outT16[(size_t)b * HDIM * SEQ + (size_t)h * SEQ + s] = (_Float16)v;
        dec_out16[i] = (_Float16)dec_out[i];
    }
}

// concat16[row, 0:512] = context[row], [512:1024] = dec_out[row]
__global__ void concat_kernel(_Float16* concat16, const float* context, const float* dec_out) {
    int n = NROW * 2 * HDIM;
    for (int i = blockIdx.x * blockDim.x + threadIdx.x; i < n; i += gridDim.x * blockDim.x) {
        int col = i & (2 * HDIM - 1);
        int row = i >> 10;
        float v = (col < HDIM) ? context[(size_t)row * HDIM + col]
                               : dec_out[(size_t)row * HDIM + (col - HDIM)];
        concat16[i] = (_Float16)v;
    }
}

// ---------------------------------------------------------------------------
// Generic batched WMMA GEMM: C[M,N] = A[M,K] * B^T + bias, B stored [N,K].
// block = 128 threads (4 waves); wave w does one 16x16 N-tile; grid = (N/64, M/16, batch).
// epilogue: 0 = none, 1 = tanh
// ---------------------------------------------------------------------------
__global__ __launch_bounds__(128)
void wmma_gemm_kernel(const _Float16* A, int lda, long long strideA,
                      const _Float16* B, int ldb, long long strideB,
                      float* C, int ldc, long long strideC,
                      int K, const float* bias, int epilogue) {
    int wave = threadIdx.x >> 5, lane = threadIdx.x & 31;
    int m0 = blockIdx.y * 16;
    int n0 = (blockIdx.x * 4 + wave) * 16;
    const _Float16* Ab = A + (size_t)blockIdx.z * strideA + (size_t)m0 * lda;
    const _Float16* Bb = B + (size_t)blockIdx.z * strideB + (size_t)n0 * ldb;
    float* Cb = C + (size_t)blockIdx.z * strideC;
    v8f acc = {};
    for (int k0 = 0; k0 < K; k0 += 32) {
        __builtin_prefetch(Bb + (size_t)(lane & 15) * ldb + k0 + 128, 0, 1);
        v16h a = load_frag(Ab, lda, lane, k0);
        v16h b = load_frag(Bb, ldb, lane, k0);
        acc = wmma_f16(a, b, acc);
    }
    int n = lane & 15, hi = (lane >> 4) & 1;
    float bv = bias ? bias[n0 + n] : 0.0f;
#pragma unroll
    for (int r = 0; r < 8; ++r) {
        float v = acc[r] + bv;
        if (epilogue == 1) v = tanhf(v);
        Cb[(size_t)(m0 + hi * 8 + r) * ldc + n0 + n] = v;
    }
}

// ---------------------------------------------------------------------------
// GRU recurrence, register-gate formulation. 3 independent blocks (enc fwd /
// enc bwd / decoder), 1024 threads = 32 waves. Wave w owns hidden columns
// j in [16w, 16w+16) for ALL three gates (Wh rows j, 512+j, 1024+j), so the
// r/z/n pre-activations for a given (row, j) land in the same lane/slot of
// three accumulators: gates evaluate entirely in registers, h_prev lives in
// registers, and only the f16 copy of h goes through LDS (double-buffered).
// Exactly one barrier per timestep.
// ---------------------------------------------------------------------------
__global__ __launch_bounds__(1024)
void gru_recur_kernel(const float* gi_f, const float* gi_b, const float* gi_d,
                      const _Float16* Wh16_f, const _Float16* Wh16_b, const _Float16* Wh16_d,
                      const float* bh_f, const float* bh_b, const float* bh_d,
                      float* out_f, float* out_b, float* out_d) {
    int g = blockIdx.x;
    const float*    gi  = (g == 0) ? gi_f  : (g == 1) ? gi_b  : gi_d;
    const _Float16* Wh  = (g == 0) ? Wh16_f: (g == 1) ? Wh16_b: Wh16_d;
    const float*    bh  = (g == 0) ? bh_f  : (g == 1) ? bh_b  : bh_d;
    float*          out = (g == 0) ? out_f : (g == 1) ? out_b : out_d;

    __shared__ _Float16 h16[2][16][HDIM];   // 32 KB, rows 8..15 stay zero

    int tid = threadIdx.x;
    for (int i = tid; i < 2 * 16 * HDIM; i += 1024) (&h16[0][0][0])[i] = (_Float16)0.0f;
    __syncthreads();

    int wave = tid >> 5, lane = tid & 31;
    int n  = lane & 15, hi = (lane >> 4) & 1;
    int j  = wave * 16 + n;                    // this lane's hidden column
    float bhr = bh[j], bhz = bh[HDIM + j], bhn = bh[2 * HDIM + j];
    const _Float16* Br = Wh + (size_t)(wave * 16) * HDIM;
    const _Float16* Bz = Wh + (size_t)(HDIM + wave * 16) * HDIM;
    const _Float16* Bn = Wh + (size_t)(2 * HDIM + wave * 16) * HDIM;

    float hprev[8] = {0.f, 0.f, 0.f, 0.f, 0.f, 0.f, 0.f, 0.f};

    for (int t = 0; t < SEQ; ++t) {
        int p = t & 1, q = p ^ 1;
        v8f ar = {}, az = {}, an = {};
        for (int k0 = 0; k0 < HDIM; k0 += 32) {
            v16h a = load_frag(&h16[p][0][0], HDIM, lane, k0);
            ar = wmma_f16(a, load_frag(Br, HDIM, lane, k0), ar);
            az = wmma_f16(a, load_frag(Bz, HDIM, lane, k0), az);
            an = wmma_f16(a, load_frag(Bn, HDIM, lane, k0), an);
        }
        int tt = (g == 1) ? (SEQ - 1 - t) : t;   // enc_b consumes/stores reversed time
        if (!hi) {                                // accumulator rows 0..7 = batch rows
#pragma unroll
            for (int r = 0; r < 8; ++r) {
                const float* gib = gi + ((size_t)r * SEQ + tt) * TH3;
                float rr = 1.0f / (1.0f + expf(-(gib[j] + ar[r] + bhr)));
                float zz = 1.0f / (1.0f + expf(-(gib[HDIM + j] + az[r] + bhz)));
                float nn = tanhf(gib[2 * HDIM + j] + rr * (an[r] + bhn));
                float hnew = (1.0f - zz) * nn + zz * hprev[r];
                hprev[r] = hnew;
                h16[q][r][j] = (_Float16)hnew;
                out[((size_t)r * SEQ + tt) * HDIM + j] = hnew;
            }
        }
        __syncthreads();    // h16[q] complete before step t+1 reads it
    }
}

// ---------------------------------------------------------------------------
// row softmax over S=256, output f16 attention weights
// ---------------------------------------------------------------------------
__global__ __launch_bounds__(256)
void softmax_kernel(const float* scores, _Float16* attn16) {
    __shared__ float red[256];
    int row = blockIdx.x, tid = threadIdx.x;
    float v = scores[(size_t)row * SEQ + tid];
    red[tid] = v; __syncthreads();
    for (int s = 128; s > 0; s >>= 1) { if (tid < s) red[tid] = fmaxf(red[tid], red[tid + s]); __syncthreads(); }
    float m = red[0]; __syncthreads();
    float e = expf(v - m);
    red[tid] = e; __syncthreads();
    for (int s = 128; s > 0; s >>= 1) { if (tid < s) red[tid] += red[tid + s]; __syncthreads(); }
    attn16[(size_t)row * SEQ + tid] = (_Float16)(e / red[0]);
}

// ---------------------------------------------------------------------------
// Output projection + online log-softmax CE, vocab-split partials.
// grid = (64 m-blocks, VSPLIT vocab splits), block = 256 threads (8 waves).
// Each block owns 32 rows (two M-tiles) and a 3200-wide vocab slice. Each wave
// loads one B fragment and feeds TWO wmma (one per M-tile): 2x FLOP per Wo
// byte out of L2. Flash-style running (max, sumexp, target-logit) per row,
// per split; merged by ce_final_kernel. Logits never hit memory.
// ---------------------------------------------------------------------------
__global__ __launch_bounds__(256)
void ce_partial_kernel(const _Float16* A16 /*[2048,512]*/, const _Float16* WoT /*[32000,512]*/,
                       const float* bo, const int* target,
                       float* pm, float* ps, float* pt) {
    __shared__ _Float16 a16[32 * HDIM];   // 32 KB
    __shared__ float    lg[32 * 128];     // 16 KB
    __shared__ float    redm[256], reds[256], redt[256];

    int mblk = blockIdx.x;                 // 0..63
    int vs   = blockIdx.y;                 // 0..VSPLIT-1
    int vbase = vs * VCHUNK;
    int tid = threadIdx.x, wave = tid >> 5, lane = tid & 31;

    for (int i = tid; i < 32 * HDIM; i += 256)
        a16[i] = A16[(size_t)mblk * 32 * HDIM + i];
    __syncthreads();

    int row = tid >> 3;                    // 0..31
    int sub = tid & 7;                     // 16 cols each
    int gr  = mblk * 32 + row;
    int tgt = target[gr];
    float m = -1e30f, s = 0.0f, tl = -1e30f;

    for (int c0 = 0; c0 < VCHUNK; c0 += 128) {
        int v0 = vbase + c0;
        int n0 = wave * 16;
        v8f acc0 = {}, acc1 = {};
        const _Float16* Bb = WoT + (size_t)(v0 + n0) * HDIM;
        for (int k0 = 0; k0 < HDIM; k0 += 32) {
            __builtin_prefetch(Bb + (size_t)(lane & 15) * HDIM + k0 + 128 * HDIM, 0, 1);
            v16h b  = load_frag(Bb, HDIM, lane, k0);
            v16h a0 = load_frag(a16, HDIM, lane, k0);
            v16h a1 = load_frag(a16 + 16 * HDIM, HDIM, lane, k0);
            acc0 = wmma_f16(a0, b, acc0);
            acc1 = wmma_f16(a1, b, acc1);
        }
        {
            int nn = lane & 15, hi = (lane >> 4) & 1;
            float bv = bo[v0 + n0 + nn];
#pragma unroll
            for (int r = 0; r < 8; ++r) {
                lg[(hi * 8 + r) * 128 + n0 + nn]        = acc0[r] + bv;
                lg[(16 + hi * 8 + r) * 128 + n0 + nn]   = acc1[r] + bv;
            }
        }
        __syncthreads();
#pragma unroll
        for (int c = 0; c < 16; ++c) {
            int col = sub * 16 + c;
            float x = lg[row * 128 + col];
            if (v0 + col == tgt) tl = x;
            float mn = fmaxf(m, x);
            s = s * expf(m - mn) + expf(x - mn);
            m = mn;
        }
        __syncthreads();
    }
    redm[tid] = m; reds[tid] = s; redt[tid] = tl;
    __syncthreads();
    if (sub == 0) {
        float M = -1e30f;
        for (int i = 0; i < 8; ++i) M = fmaxf(M, redm[row * 8 + i]);
        float S = 0.0f, TL = -1e30f;
        for (int i = 0; i < 8; ++i) {
            S  += reds[row * 8 + i] * expf(redm[row * 8 + i] - M);
            TL  = fmaxf(TL, redt[row * 8 + i]);
        }
        pm[(size_t)gr * VSPLIT + vs] = M;
        ps[(size_t)gr * VSPLIT + vs] = S;
        pt[(size_t)gr * VSPLIT + vs] = TL;
    }
}

// merge vocab-split partials, compute per-row nll, reduce to final scalar
__global__ __launch_bounds__(256)
void ce_final_kernel(const float* pm, const float* ps, const float* pt, float* out) {
    __shared__ float red[256];
    float acc = 0.0f;
    for (int gr = threadIdx.x; gr < NROW; gr += 256) {
        float M = -1e30f;
        for (int i = 0; i < VSPLIT; ++i) M = fmaxf(M, pm[(size_t)gr * VSPLIT + i]);
        float S = 0.0f, TL = -1e30f;
        for (int i = 0; i < VSPLIT; ++i) {
            S  += ps[(size_t)gr * VSPLIT + i] * expf(pm[(size_t)gr * VSPLIT + i] - M);
            TL  = fmaxf(TL, pt[(size_t)gr * VSPLIT + i]);
        }
        acc += (M + logf(S)) - TL;        // -log_softmax at target
    }
    red[threadIdx.x] = acc; __syncthreads();
    for (int k = 128; k > 0; k >>= 1) { if (threadIdx.x < k) red[threadIdx.x] += red[threadIdx.x + k]; __syncthreads(); }
    if (threadIdx.x == 0) out[0] = red[0] / (float)SEQ;  // mean over T, summed over B
}

// ---------------------------------------------------------------------------
extern "C" void kernel_launch(void* const* d_in, const int* in_sizes, int n_in,
                              void* d_out, int out_size, void* d_ws, size_t ws_size,
                              hipStream_t stream) {
    const int*   input_seq  = (const int*)d_in[0];
    const int*   target_seq = (const int*)d_in[1];
    const float* embedding  = (const float*)d_in[2];
    const float* Wi_f = (const float*)d_in[3];  const float* Wh_f = (const float*)d_in[4];
    const float* bi_f = (const float*)d_in[5];  const float* bh_f = (const float*)d_in[6];
    const float* Wi_b = (const float*)d_in[7];  const float* Wh_b = (const float*)d_in[8];
    const float* bi_b = (const float*)d_in[9];  const float* bh_b = (const float*)d_in[10];
    const float* Wi_d = (const float*)d_in[11]; const float* Wh_d = (const float*)d_in[12];
    const float* bi_d = (const float*)d_in[13]; const float* bh_d = (const float*)d_in[14];
    const float* Wc   = (const float*)d_in[15];
    const float* Wo   = (const float*)d_in[16];
    const float* bo   = (const float*)d_in[17];
    float* out = (float*)d_out;

    char* wp = (char*)d_ws;
    auto alloc = [&](size_t bytes) -> void* {
        void* p = (void*)wp;
        wp += (bytes + 255) & ~(size_t)255;
        return p;
    };

    _Float16* enc_emb16 = (_Float16*)alloc((size_t)NROW * HDIM * 2);
    _Float16* dec_emb16 = (_Float16*)alloc((size_t)NROW * HDIM * 2);
    _Float16* Wi16_f = (_Float16*)alloc((size_t)TH3 * HDIM * 2);
    _Float16* Wh16_f = (_Float16*)alloc((size_t)TH3 * HDIM * 2);
    _Float16* Wi16_b = (_Float16*)alloc((size_t)TH3 * HDIM * 2);
    _Float16* Wh16_b = (_Float16*)alloc((size_t)TH3 * HDIM * 2);
    _Float16* Wi16_d = (_Float16*)alloc((size_t)TH3 * HDIM * 2);
    _Float16* Wh16_d = (_Float16*)alloc((size_t)TH3 * HDIM * 2);
    float* gi_f = (float*)alloc((size_t)NROW * TH3 * 4);
    float* gi_b = (float*)alloc((size_t)NROW * TH3 * 4);
    float* gi_d = (float*)alloc((size_t)NROW * TH3 * 4);
    float* enc_f_out = (float*)alloc((size_t)NROW * HDIM * 4);
    float* enc_b_out = (float*)alloc((size_t)NROW * HDIM * 4);
    float* dec_out   = (float*)alloc((size_t)NROW * HDIM * 4);
    float* enc_out   = (float*)alloc((size_t)NROW * HDIM * 4);
    _Float16* enc_out16  = (_Float16*)alloc((size_t)NROW * HDIM * 2);
    _Float16* enc_outT16 = (_Float16*)alloc((size_t)NROW * HDIM * 2);
    _Float16* dec_out16  = (_Float16*)alloc((size_t)NROW * HDIM * 2);
    float*    scores  = (float*)alloc((size_t)BB * SEQ * SEQ * 4);
    _Float16* attn16  = (_Float16*)alloc((size_t)BB * SEQ * SEQ * 2);
    float*    context = (float*)alloc((size_t)NROW * HDIM * 4);
    _Float16* concat16 = (_Float16*)alloc((size_t)NROW * 2 * HDIM * 2);
    float*    att_out = (float*)alloc((size_t)NROW * HDIM * 4);
    _Float16* attout16 = (_Float16*)alloc((size_t)NROW * HDIM * 2);
    _Float16* WcT16 = (_Float16*)alloc((size_t)HDIM * (2 * HDIM) * 2);
    _Float16* WoT16 = (_Float16*)alloc((size_t)VOC * HDIM * 2);
    float* pm = (float*)alloc((size_t)NROW * VSPLIT * 4);
    float* ps = (float*)alloc((size_t)NROW * VSPLIT * 4);
    float* pt = (float*)alloc((size_t)NROW * VSPLIT * 4);

    const int wN = TH3 * HDIM;
    cvt_f32_f16_kernel<<<1024, 256, 0, stream>>>(Wi16_f, Wi_f, wN);
    cvt_f32_f16_kernel<<<1024, 256, 0, stream>>>(Wh16_f, Wh_f, wN);
    cvt_f32_f16_kernel<<<1024, 256, 0, stream>>>(Wi16_b, Wi_b, wN);
    cvt_f32_f16_kernel<<<1024, 256, 0, stream>>>(Wh16_b, Wh_b, wN);
    cvt_f32_f16_kernel<<<1024, 256, 0, stream>>>(Wi16_d, Wi_d, wN);
    cvt_f32_f16_kernel<<<1024, 256, 0, stream>>>(Wh16_d, Wh_d, wN);
    transpose_f32_f16_kernel<<<2048, 256, 0, stream>>>(WcT16, Wc, 2 * HDIM, HDIM);
    transpose_f32_f16_kernel<<<8192, 256, 0, stream>>>(WoT16, Wo, HDIM, VOC);
    gather_emb_kernel<<<4096, 256, 0, stream>>>(enc_emb16, dec_emb16, embedding,
                                                input_seq, target_seq);

    // gi = x @ Wi^T + bi for all timesteps (M=2048, N=1536, K=512)
    {
        dim3 grid(TH3 / 64, NROW / 16, 1);
        wmma_gemm_kernel<<<grid, 128, 0, stream>>>(enc_emb16, HDIM, 0, Wi16_f, HDIM, 0,
                                                   gi_f, TH3, 0, HDIM, bi_f, 0);
        wmma_gemm_kernel<<<grid, 128, 0, stream>>>(enc_emb16, HDIM, 0, Wi16_b, HDIM, 0,
                                                   gi_b, TH3, 0, HDIM, bi_b, 0);
        wmma_gemm_kernel<<<grid, 128, 0, stream>>>(dec_emb16, HDIM, 0, Wi16_d, HDIM, 0,
                                                   gi_d, TH3, 0, HDIM, bi_d, 0);
    }

    // sequential GRU recurrences (3 independent blocks, register-gate form)
    gru_recur_kernel<<<3, 1024, 0, stream>>>(gi_f, gi_b, gi_d,
                                             Wh16_f, Wh16_b, Wh16_d,
                                             bh_f, bh_b, bh_d,
                                             enc_f_out, enc_b_out, dec_out);

    combine_kernel<<<4096, 256, 0, stream>>>(enc_f_out, enc_b_out, dec_out,
                                             enc_out, enc_out16, enc_outT16, dec_out16);

    // scores[b] = dec_out[b] @ enc_out[b]^T  (batched M=256, N=256, K=512)
    {
        dim3 grid(SEQ / 64, SEQ / 16, BB);
        wmma_gemm_kernel<<<grid, 128, 0, stream>>>(dec_out16, HDIM, (long long)SEQ * HDIM,
                                                   enc_out16, HDIM, (long long)SEQ * HDIM,
                                                   scores, SEQ, (long long)SEQ * SEQ,
                                                   HDIM, nullptr, 0);
    }
    softmax_kernel<<<NROW, 256, 0, stream>>>(scores, attn16);

    // context[b] = attn[b] @ enc_out[b]  (batched M=256, N=512, K=256; B = enc_outT)
    {
        dim3 grid(HDIM / 64, SEQ / 16, BB);
        wmma_gemm_kernel<<<grid, 128, 0, stream>>>(attn16, SEQ, (long long)SEQ * SEQ,
                                                   enc_outT16, SEQ, (long long)HDIM * SEQ,
                                                   context, HDIM, (long long)SEQ * HDIM,
                                                   SEQ, nullptr, 0);
    }

    concat_kernel<<<8192, 256, 0, stream>>>(concat16, context, dec_out);

    // att_out = tanh(concat @ Wc)  (M=2048, N=512, K=1024)
    {
        dim3 grid(HDIM / 64, NROW / 16, 1);
        wmma_gemm_kernel<<<grid, 128, 0, stream>>>(concat16, 2 * HDIM, 0,
                                                   WcT16, 2 * HDIM, 0,
                                                   att_out, HDIM, 0,
                                                   2 * HDIM, nullptr, 1);
    }
    cvt_f32_f16_kernel<<<4096, 256, 0, stream>>>(attout16, att_out, NROW * HDIM);

    // fused output projection + online log-softmax CE (dominant 67 GF GEMM)
    {
        dim3 grid(NROW / 32, VSPLIT);
        ce_partial_kernel<<<grid, 256, 0, stream>>>(attout16, WoT16, bo, target_seq,
                                                    pm, ps, pt);
    }
    ce_final_kernel<<<1, 256, 0, stream>>>(pm, ps, pt, out);

    (void)in_sizes; (void)n_in; (void)out_size; (void)ws_size;
}
